// LITLayer_69054484185187
// MI455X (gfx1250) — matrix operand order repaired and checked by reading, not compiled, not run on hardware
//
#include <hip/hip_runtime.h>
#include <hip/hip_bf16.h>
#include <math.h>

typedef __attribute__((ext_vector_type(16))) _Float16 v16h;
typedef __attribute__((ext_vector_type(8)))  _Float16 v8h;
typedef __attribute__((ext_vector_type(8)))  float    v8f;

#define Bn     2
#define DIM    128
#define Hh     64
#define Ww     64
#define HEADS  4
#define Rr     3
#define RA     49
#define HRr    96
#define Qq     (HRr*HRr)      // 9216
#define NQ     (Bn*Qq)        // 18432
#define HIDDEN 256
#define KPAD   160            // 130 padded to 5*32

static __device__ __forceinline__ v16h mkv16(const _Float16* p) {
  v8h lo = *(const v8h*)p;
  v8h hi = *(const v8h*)(p + 16);
  v16h a;
#pragma unroll
  for (int e = 0; e < 8; ++e) { a[e] = lo[e]; a[e + 8] = hi[e]; }
  return a;
}

// ---------------- NCHW -> NHWC transpose of input ----------------
__global__ __launch_bounds__(256) void k_nchw_to_nhwc(const float* __restrict__ in,
                                                      float* __restrict__ out) {
  int idx = blockIdx.x * 256 + threadIdx.x;   // total Bn*DIM*Hh*Ww = 2^20
  int x  = idx & 63;
  int y  = (idx >> 6) & 63;
  int ch = (idx >> 12) & 127;
  int b  = idx >> 19;
  out[(((b * Hh + y) * Ww + x) * DIM) + ch] = in[idx];
}

// ------------- pack conv weights into per-lane WMMA B fragments -------------
// layout: [c][tap][ntile][kc][lane][16 halfs]; B[k][n]: n=lane&15, k=16*(h>>3)+8*(lane>>4)+(h&7)
__global__ __launch_bounds__(256) void k_pack_convw(const float* __restrict__ qw,
                                                    const float* __restrict__ kw,
                                                    const float* __restrict__ vw,
                                                    _Float16* __restrict__ wfrag) {
  int idx = blockIdx.x * 256 + threadIdx.x;   // total 3*9*8*4*32*16 = 442368
  int h  = idx & 15;
  int l  = (idx >> 4) & 31;
  int kc = (idx >> 9) & 3;
  int nt = (idx >> 11) & 7;
  int t  = (idx >> 14) % 9;
  int c  = idx / (16 * 32 * 4 * 8 * 9);
  const float* w = (c == 0) ? qw : ((c == 1) ? kw : vw);
  int o = nt * 16 + (l & 15);
  int i = kc * 32 + ((h >> 3) << 4) + ((l >> 4) << 3) + (h & 7);
  int ky = t / 3, kx = t % 3;
  wfrag[idx] = (_Float16)w[((o * DIM + i) * 3 + ky) * 3 + kx];
}

// ------------- pack MLP weights (w0 K-padded 130->160, w1) -------------
__global__ __launch_bounds__(256) void k_pack_mlpw(const float* __restrict__ w0,
                                                   const float* __restrict__ w1,
                                                   _Float16* __restrict__ w0f,
                                                   _Float16* __restrict__ w1f) {
  int idx = blockIdx.x * 256 + threadIdx.x;   // total 40960 + 32768 = 73728
  if (idx < 16 * 5 * 512) {
    int h = idx & 15, l = (idx >> 4) & 31;
    int kc = (idx >> 9) % 5, nt = (idx >> 9) / 5;
    int n = nt * 16 + (l & 15);
    int k = kc * 32 + ((h >> 3) << 4) + ((l >> 4) << 3) + (h & 7);
    w0f[idx] = (_Float16)((k < DIM + 2) ? w0[n * (DIM + 2) + k] : 0.0f);
  } else {
    int j = idx - 16 * 5 * 512;
    int h = j & 15, l = (j >> 4) & 31;
    int kc = (j >> 9) & 7, nt = (j >> 9) >> 3;
    int n = nt * 16 + (l & 15);
    int k = kc * 32 + ((h >> 3) << 4) + ((l >> 4) << 3) + (h & 7);
    w1f[j] = (_Float16)w1[n * HIDDEN + k];
  }
}

// ------------- conv3x3 (q,k,v) as implicit GEMM over 9 taps, WMMA f16 -------------
// grid: (Bn*Hh, Ww/16, 3); block 256 (8 waves). Wave=out-channel tile, 16 pixels/tile.
__global__ __launch_bounds__(256) void k_conv3x3_qkv(const float* __restrict__ xt,
                                                     const _Float16* __restrict__ wfrag,
                                                     const float* __restrict__ qb,
                                                     const float* __restrict__ kb,
                                                     const float* __restrict__ vb,
                                                     float* __restrict__ fq,
                                                     float* __restrict__ fk,
                                                     float* __restrict__ fv) {
  __shared__ __align__(16) _Float16 smem[3 * 18 * DIM];  // rows y-1..y+1, cols x0-1..x0+16
  int tid = threadIdx.x;
  int by = blockIdx.x;
  int b = by >> 6, y = by & 63;
  int x0 = blockIdx.y * 16;
  int c = blockIdx.z;

  for (int t = tid; t < 3 * 18 * DIM; t += 256) {
    int ch = t & 127;
    int cc = (t >> 7) % 18;
    int r  = (t >> 7) / 18;
    int gy = y + r - 1;
    int gx = x0 + cc - 1;
    float v = 0.0f;
    if (gy >= 0 && gy < Hh && gx >= 0 && gx < Ww)
      v = xt[(((size_t)b * Hh + gy) * Ww + gx) * DIM + ch];
    smem[t] = (_Float16)v;
  }
  __syncthreads();

  int wave = tid >> 5, lane = tid & 31;
  int m = lane & 15, ksel = lane >> 4;
  int nt = wave;
  v8f acc = {};
  const _Float16* wbase = wfrag + (size_t)c * 9 * 8 * 4 * 512;
#pragma unroll
  for (int t = 0; t < 9; ++t) {
    int ky = t / 3, kx = t % 3;
    const _Float16* ap = &smem[((ky * 18) + (m + kx)) * DIM];
#pragma unroll
    for (int kc = 0; kc < 4; ++kc) {
      v16h a = mkv16(ap + kc * 32 + 8 * ksel);
      v16h bm = *(const v16h*)(wbase + (((t * 8 + nt) * 4 + kc) << 9) + lane * 16);
      acc = __builtin_amdgcn_wmma_f32_16x16x32_f16(false, a, false, bm, (short)0, acc,
                                                   false, false);
    }
  }
  int och = nt * 16 + (lane & 15);
  const float* bias = (c == 0) ? qb : ((c == 1) ? kb : vb);
  float* out = (c == 0) ? fq : ((c == 1) ? fk : fv);
  float bv = bias[och];
#pragma unroll
  for (int r = 0; r < 8; ++r) {
    int xx = x0 + r + 8 * ksel;
    out[(((size_t)b * Hh + y) * Ww + xx) * DIM + och] = acc[r] + bv;
  }
}

// ------------- local attention + softmax + skip; one wave per query -------------
// Lane owns channels {lane, lane+32, lane+64, lane+96} (one per head).
__global__ __launch_bounds__(256) void k_attn(const float* __restrict__ coord,
                                              const float* __restrict__ cell,
                                              const float* __restrict__ xt,
                                              const float* __restrict__ fq,
                                              const float* __restrict__ fk,
                                              const float* __restrict__ fv,
                                              const float* __restrict__ pb_w,
                                              const float* __restrict__ pb_b,
                                              _Float16* __restrict__ Xf,
                                              float* __restrict__ skip) {
  __shared__ float lw[8][RA * HEADS];
  int wave = threadIdx.x >> 5, lane = threadIdx.x & 31;
  int qi = blockIdx.x * 8 + wave;         // 2304*8 = 18432 exactly
  int b = qi / Qq;
  float cy = coord[(size_t)qi * 2 + 0];
  float cx = coord[(size_t)qi * 2 + 1];

  float py = ((cy + 1.0f) * Hh - 1.0f) * 0.5f;
  float px = ((cx + 1.0f) * Ww - 1.0f) * 0.5f;
  // nearest-sample of coord_map (zeros padding) -> center coords sck0
  int iy0 = (int)rintf(py), ix0 = (int)rintf(px);
  bool inb0 = (iy0 >= 0 && iy0 < Hh && ix0 >= 0 && ix0 < Ww);
  int iy0c = min(max(iy0, 0), Hh - 1), ix0c = min(max(ix0, 0), Ww - 1);
  float sck0y = inb0 ? (-1.0f + (2.0f * iy0c + 1.0f) / Hh) : 0.0f;
  float sck0x = inb0 ? (-1.0f + (2.0f * ix0c + 1.0f) / Ww) : 0.0f;

  int ch[4];
#pragma unroll
  for (int i = 0; i < 4; ++i) ch[i] = lane + 32 * i;

  // bilinear (zeros) sample of feat_q at sc
  float qv[4] = {0, 0, 0, 0};
  {
    float fy = floorf(py), fx = floorf(px);
    float wy1 = py - fy, wx1 = px - fx;
    int y0i = (int)fy, x0i = (int)fx;
#pragma unroll
    for (int dy = 0; dy < 2; ++dy)
#pragma unroll
      for (int dx = 0; dx < 2; ++dx) {
        int iy = y0i + dy, ix = x0i + dx;
        float wgt = (dy ? wy1 : 1.0f - wy1) * (dx ? wx1 : 1.0f - wx1);
        if (iy < 0 || iy >= Hh || ix < 0 || ix >= Ww) wgt = 0.0f;
        int iyc = min(max(iy, 0), Hh - 1), ixc = min(max(ix, 0), Ww - 1);
        const float* p = fq + (((size_t)b * Hh + iyc) * Ww + ixc) * DIM;
#pragma unroll
        for (int i = 0; i < 4; ++i) qv[i] += wgt * p[ch[i]];
      }
  }
  const float inv_sqrt_dk = 0.17677669529663687f;  // 1/sqrt(32)
#pragma unroll
  for (int i = 0; i < 4; ++i) qv[i] *= inv_sqrt_dk;

  // positional-bias constants: lane == encoding index (0..31)
  float freq = exp2f((float)(lane & 15) * (10.0f / 15.0f)) - 1.0f;
  float pbwA[HEADS], pbwB[HEADS];
#pragma unroll
  for (int h = 0; h < HEADS; ++h) {
    pbwA[h] = pb_w[h * 64 + lane];
    pbwB[h] = pb_w[h * 64 + 32 + lane];
  }
  float pbb = pb_b[lane & 3];
  const float rh = 2.0f / Hh, rw = 2.0f / Ww;

  for (int tap = 0; tap < RA; ++tap) {
    int dyi = tap / 7 - Rr, dxi = tap % 7 - Rr;
    float scky = sck0y + dyi * rh;
    float sckx = sck0x + dxi * rw;
    int iyk = (int)rintf(((scky + 1.0f) * Hh - 1.0f) * 0.5f);
    int ixk = (int)rintf(((sckx + 1.0f) * Ww - 1.0f) * 0.5f);
    bool inbk = (iyk >= 0 && iyk < Hh && ixk >= 0 && ixk < Ww);
    int iykc = min(max(iyk, 0), Hh - 1), ixkc = min(max(ixk, 0), Ww - 1);
    const float* kp = fk + (((size_t)b * Hh + iykc) * Ww + ixkc) * DIM;
    float p4[HEADS];
#pragma unroll
    for (int i = 0; i < HEADS; ++i) p4[i] = inbk ? qv[i] * kp[ch[i]] : 0.0f;
    // fold pos-bias into the same cross-lane reduction
    float rcy = (cy - scky) * Hh, rcx = (cx - sckx) * Ww;
    float proj = (lane < 16 ? rcy : rcx) * freq;
    float sn = __sinf(proj), cs = __cosf(proj);
#pragma unroll
    for (int h = 0; h < HEADS; ++h) p4[h] += pbwA[h] * sn + pbwB[h] * cs;
#pragma unroll
    for (int off = 16; off >= 1; off >>= 1)
#pragma unroll
      for (int h = 0; h < HEADS; ++h) p4[h] += __shfl_xor(p4[h], off, 32);
    if (lane < HEADS) lw[wave][tap * HEADS + lane] = p4[lane] + pbb;
  }
  __syncthreads();

  if (lane < HEADS) {
    float mx = -1e30f;
    for (int t = 0; t < RA; ++t) mx = fmaxf(mx, lw[wave][t * HEADS + lane]);
    float s = 0.0f;
    for (int t = 0; t < RA; ++t) s += __expf(lw[wave][t * HEADS + lane] - mx);
    float inv = 1.0f / s;
    for (int t = 0; t < RA; ++t)
      lw[wave][t * HEADS + lane] = __expf(lw[wave][t * HEADS + lane] - mx) * inv;
  }
  __syncthreads();

  float agg[4] = {0, 0, 0, 0};
  for (int tap = 0; tap < RA; ++tap) {
    int dyi = tap / 7 - Rr, dxi = tap % 7 - Rr;
    float scky = sck0y + dyi * rh;
    float sckx = sck0x + dxi * rw;
    int iyk = (int)rintf(((scky + 1.0f) * Hh - 1.0f) * 0.5f);
    int ixk = (int)rintf(((sckx + 1.0f) * Ww - 1.0f) * 0.5f);
    bool inbk = (iyk >= 0 && iyk < Hh && ixk >= 0 && ixk < Ww);
    int iykc = min(max(iyk, 0), Hh - 1), ixkc = min(max(ixk, 0), Ww - 1);
    const float* vp = fv + (((size_t)b * Hh + iykc) * Ww + ixkc) * DIM;
#pragma unroll
    for (int i = 0; i < 4; ++i) {
      float val = inbk ? vp[ch[i]] : 0.0f;
      agg[i] += lw[wave][tap * HEADS + i] * val;
    }
  }

  // skip: bilinear (border) sample of the raw input (NHWC copy)
  float sk[4] = {0, 0, 0, 0};
  {
    float pyb = fminf(fmaxf(py, 0.0f), (float)(Hh - 1));
    float pxb = fminf(fmaxf(px, 0.0f), (float)(Ww - 1));
    float fy = floorf(pyb), fx = floorf(pxb);
    float wy1 = pyb - fy, wx1 = pxb - fx;
    int y0i = (int)fy, x0i = (int)fx;
#pragma unroll
    for (int dy = 0; dy < 2; ++dy)
#pragma unroll
      for (int dx = 0; dx < 2; ++dx) {
        int iy = min(max(y0i + dy, 0), Hh - 1);
        int ix = min(max(x0i + dx, 0), Ww - 1);
        float wgt = (dy ? wy1 : 1.0f - wy1) * (dx ? wx1 : 1.0f - wx1);
        const float* p = xt + (((size_t)b * Hh + iy) * Ww + ix) * DIM;
#pragma unroll
        for (int i = 0; i < 4; ++i) sk[i] += wgt * p[ch[i]];
      }
  }
  size_t row = (size_t)qi;
#pragma unroll
  for (int i = 0; i < 4; ++i) skip[row * DIM + ch[i]] = sk[i];
  _Float16* xr = Xf + row * KPAD;
#pragma unroll
  for (int i = 0; i < 4; ++i) xr[ch[i]] = (_Float16)agg[i];
  if (lane == 0) {
    xr[128] = (_Float16)(cell[b * 2 + 0] * (float)Hh);
    xr[129] = (_Float16)(cell[b * 2 + 1] * (float)Ww);
  }
  if (lane < 30) xr[130 + lane] = (_Float16)0.0f;
}

// ------------- fused MLP: GEMM1 + exact GELU + GEMM2 + bias + skip -------------
// grid 1152 (16 rows each), block 256 (8 waves). GEMM1: wave handles 2 N-tiles of 16.
__global__ __launch_bounds__(256) void k_mlp(const _Float16* __restrict__ Xf,
                                             const _Float16* __restrict__ w0f,
                                             const float* __restrict__ b0,
                                             const _Float16* __restrict__ w1f,
                                             const float* __restrict__ b1,
                                             const float* __restrict__ skip,
                                             float* __restrict__ out) {
  __shared__ __align__(16) _Float16 hid[16 * HIDDEN];
  int wave = threadIdx.x >> 5, lane = threadIdx.x & 31;
  int m = lane & 15, ksel = lane >> 4;
  int bq0 = blockIdx.x * 16;
  __builtin_prefetch(w1f + ((wave * 8) << 9) + lane * 16, 0, 1);

#pragma unroll
  for (int sub = 0; sub < 2; ++sub) {
    int nt = wave * 2 + sub;
    v8f acc = {};
#pragma unroll
    for (int kc = 0; kc < 5; ++kc) {
      v16h a = mkv16(Xf + (size_t)(bq0 + m) * KPAD + kc * 32 + 8 * ksel);
      v16h bm = *(const v16h*)(w0f + ((nt * 5 + kc) << 9) + lane * 16);
      acc = __builtin_amdgcn_wmma_f32_16x16x32_f16(false, a, false, bm, (short)0, acc,
                                                   false, false);
    }
    int n = nt * 16 + (lane & 15);
    float bb = b0[n];
#pragma unroll
    for (int r = 0; r < 8; ++r) {
      int mm = r + 8 * ksel;
      float xv = acc[r] + bb;
      float g = 0.5f * xv * (1.0f + erff(xv * 0.70710678118654752f));  // exact GELU
      hid[mm * HIDDEN + n] = (_Float16)g;
    }
  }
  __syncthreads();

  {
    int nt = wave;
    v8f acc = {};
#pragma unroll
    for (int kc = 0; kc < 8; ++kc) {
      v16h a = mkv16(hid + m * HIDDEN + kc * 32 + 8 * ksel);
      v16h bm = *(const v16h*)(w1f + ((nt * 8 + kc) << 9) + lane * 16);
      acc = __builtin_amdgcn_wmma_f32_16x16x32_f16(false, a, false, bm, (short)0, acc,
                                                   false, false);
    }
    int chn = nt * 16 + (lane & 15);
    float bb = b1[chn];
#pragma unroll
    for (int r = 0; r < 8; ++r) {
      int mm = r + 8 * ksel;
      size_t row = (size_t)(bq0 + mm);
      out[row * DIM + chn] = acc[r] + bb + skip[row * DIM + chn];
    }
  }
}

extern "C" void kernel_launch(void* const* d_in, const int* in_sizes, int n_in,
                              void* d_out, int out_size, void* d_ws, size_t ws_size,
                              hipStream_t stream) {
  (void)in_sizes; (void)n_in; (void)out_size; (void)ws_size;
  const float* x     = (const float*)d_in[0];
  const float* coord = (const float*)d_in[1];
  const float* cell  = (const float*)d_in[3];
  const float* qw    = (const float*)d_in[4];
  const float* qb    = (const float*)d_in[5];
  const float* kw    = (const float*)d_in[6];
  const float* kb    = (const float*)d_in[7];
  const float* vw    = (const float*)d_in[8];
  const float* vb    = (const float*)d_in[9];
  const float* pb_w  = (const float*)d_in[10];
  const float* pb_b  = (const float*)d_in[11];
  const float* w0    = (const float*)d_in[12];
  const float* b0    = (const float*)d_in[13];
  const float* w1    = (const float*)d_in[14];
  const float* b1    = (const float*)d_in[15];

  char* ws = (char*)d_ws;
  float*    xt    = (float*)(ws + 0);                  //  4 MB  NHWC input
  float*    fq    = (float*)(ws + 4194304);            //  4 MB
  float*    fk    = (float*)(ws + 8388608);            //  4 MB
  float*    fv    = (float*)(ws + 12582912);           //  4 MB
  _Float16* wfrag = (_Float16*)(ws + 16777216);        //  884736 B
  _Float16* w0f   = (_Float16*)(ws + 17661952);        //  81920 B
  _Float16* w1f   = (_Float16*)(ws + 17743872);        //  65536 B
  _Float16* Xf    = (_Float16*)(ws + 17809408);        //  5898240 B (18432 x 160 f16)
  float*    skipb = (float*)(ws + 23707648);           //  9437184 B
  float*    outp  = (float*)d_out;                     //  (B,Q,128) f32

  k_nchw_to_nhwc<<<8192, 256, 0, stream>>>(x, xt);
  k_pack_convw<<<1728, 256, 0, stream>>>(qw, kw, vw, wfrag);
  k_pack_mlpw<<<288, 256, 0, stream>>>(w0, w1, w0f, w1f);
  k_conv3x3_qkv<<<dim3(Bn * Hh, Ww / 16, 3), 256, 0, stream>>>(xt, wfrag, qb, kb, vb,
                                                               fq, fk, fv);
  k_attn<<<NQ / 8, 256, 0, stream>>>(coord, cell, xt, fq, fk, fv, pb_w, pb_b, Xf, skipb);
  k_mlp<<<NQ / 16, 256, 0, stream>>>(Xf, w0f, b0, w1f, b1, skipb, outp);
}
